// Transformer_38379827757769
// MI455X (gfx1250) — matrix operand order repaired and checked by reading, not compiled, not run on hardware
//
#include <hip/hip_runtime.h>
#include <math.h>

// ---------------- constants ----------------
#define TOK   1026        // B*(S+1) = 2*513
#define SEQ   513
#define DIMC  1024
#define NH    16
#define HD    64
#define HID   2816
#define UPN   5632
#define QKVN  3072
#define NEXP  8
#define VOC   16384
#define MAXR  2052        // max rows per expert list (TOK*TOP_K)

// ---------------- WMMA types ----------------
typedef __attribute__((ext_vector_type(16))) __bf16 v16bf;
typedef __attribute__((ext_vector_type(8)))  float  v8f;
typedef __attribute__((ext_vector_type(4)))  float  v4f;
typedef __attribute__((ext_vector_type(4)))  __bf16 v4bf;

__device__ __forceinline__ v8f zero8() {
  v8f z = {0.f,0.f,0.f,0.f,0.f,0.f,0.f,0.f};
  return z;
}

// 4x fp32 -> 4x bf16 (one vector fptrunc -> two v_cvt_pk_bf16_f32), packed in 8 bytes
__device__ __forceinline__ unsigned long long f4_to_bf4(float4 v) {
  union { v4bf b; unsigned long long u; } c;
  v4f f;
  f.x = v.x; f.y = v.y; f.z = v.z; f.w = v.w;
  c.b = __builtin_convertvector(f, v4bf);
  return c.u;
}

// A fragment: 16x32 bf16, per ISA table (VGPR j<4: K=2j+8*half; j>=4: K=16+2(j-4)+8*half)
__device__ __forceinline__ v16bf load_a_frag(const unsigned short (*As)[36], int mb, int lane) {
  union { v16bf v; unsigned short u[16]; } a;
  int hl = lane >> 4;
  int m  = mb + (lane & 15);
  #pragma unroll
  for (int e = 0; e < 16; ++e) {
    int j = e >> 1, p = e & 1;
    int k = ((j >> 2) << 4) + (hl << 3) + ((j & 3) << 1) + p;
    a.u[e] = As[m][k];
  }
  return a.v;
}

// B fragment: 32x16 bf16, same K order as A (e<8: K=8*half+e; e>=8: K=16+8*half+(e-8)) (N = lane&15)
__device__ __forceinline__ v16bf load_b_frag(const unsigned short (*Bs)[36], int nb, int lane) {
  union { v16bf v; unsigned short u[16]; } b;
  int hl = lane >> 4;
  int n  = nb + (lane & 15);
  #pragma unroll
  for (int e = 0; e < 16; ++e) b.u[e] = Bs[n][((e >> 3) << 4) + (hl << 3) + (e & 7)];
  return b.v;
}

// ---------------- block reductions ----------------
__device__ __forceinline__ float blk_reduce_sum(float v, float* red) {
  int tid = threadIdx.x, lane = tid & 31;
  #pragma unroll
  for (int o = 16; o > 0; o >>= 1) v += __shfl_xor(v, o, 32);
  __syncthreads();
  if (lane == 0) red[tid >> 5] = v;
  __syncthreads();
  float t = (tid < (int)(blockDim.x >> 5)) ? red[tid] : 0.f;
  #pragma unroll
  for (int o = 16; o > 0; o >>= 1) t += __shfl_xor(t, o, 32);
  __syncthreads();
  if (tid == 0) red[0] = t;
  __syncthreads();
  return red[0];
}

__device__ __forceinline__ float blk_reduce_max(float v, float* red) {
  int tid = threadIdx.x, lane = tid & 31;
  #pragma unroll
  for (int o = 16; o > 0; o >>= 1) v = fmaxf(v, __shfl_xor(v, o, 32));
  __syncthreads();
  if (lane == 0) red[tid >> 5] = v;
  __syncthreads();
  float t = (tid < (int)(blockDim.x >> 5)) ? red[tid] : -3.0e38f;
  #pragma unroll
  for (int o = 16; o > 0; o >>= 1) t = fmaxf(t, __shfl_xor(t, o, 32));
  __syncthreads();
  if (tid == 0) red[0] = t;
  __syncthreads();
  return red[0];
}

// ---------------- RoPE tables (derived constants) ----------------
__global__ void rope_init_kernel(float* __restrict__ cosT, float* __restrict__ sinT) {
  int i = blockIdx.x * blockDim.x + threadIdx.x;
  if (i >= SEQ * 16) return;
  int p = i >> 4, f = i & 15;
  float fr = 3.14159265358979f * expf((float)(f & 7) * (2.302585092994046f / 7.0f));
  float pv = 0.f;
  if (p > 0) {
    int g = p - 1, gi = g >> 5, gj = g & 31;
    float mi = (2.f * gi - 31.f) / 32.f;
    float mj = (2.f * gj - 31.f) / 32.f;
    pv = (f < 8) ? mi : mj;
  }
  float th = pv * fr;
  cosT[i] = cosf(th);
  sinT[i] = sinf(th);
}

// ---------------- embedding: h = [class_embed[y], expert_w[0][x]] ----------------
__global__ void embed_kernel(const int* __restrict__ x, const int* __restrict__ y,
                             const float* __restrict__ cls, const float* __restrict__ ew0,
                             float* __restrict__ h) {
  int i = blockIdx.x * blockDim.x + threadIdx.x;
  if (i >= TOK * DIMC) return;
  int c = i & (DIMC - 1);
  int row = i >> 10;
  int b = row / SEQ, s = row % SEQ;
  float v;
  if (s == 0) v = cls[(size_t)y[b] * DIMC + c];
  else        v = ew0[(size_t)x[b * (SEQ - 1) + (s - 1)] * DIMC + c];
  h[i] = v;
}

// ---------------- RMSNorm (1024 cols, one block per row) ----------------
__global__ __launch_bounds__(256) void rmsnorm_kernel(const float* __restrict__ x,
                                                      const float* __restrict__ g,
                                                      float* __restrict__ o) {
  __shared__ float red[8];
  int row = blockIdx.x, tid = threadIdx.x;
  const float* xr = x + (size_t)row * DIMC;
  float ss = 0.f;
  for (int c = tid; c < DIMC; c += 256) { float v = xr[c]; ss += v * v; }
  float tot = blk_reduce_sum(ss, red);
  float inv = rsqrtf(tot * (1.0f / DIMC) + 1e-6f);
  float* orow = o + (size_t)row * DIMC;
  for (int c = tid; c < DIMC; c += 256) orow[c] = xr[c] * g[c] * inv;
}

// ---------------- tiled bf16 WMMA GEMM: C[M,N] = A[M,K] @ W[N,K]^T (+R) ----------------
// Block tile 64(M) x 128(N), 4 waves, each wave 32x64 = 8 WMMAs per K-step.
// Global loads for step k+1 are register-staged while WMMAs of step k execute.
__global__ __launch_bounds__(128) void gemm_kernel(const float* __restrict__ A,
                                                   const float* __restrict__ W,
                                                   const float* __restrict__ R,
                                                   float* __restrict__ C,
                                                   int M, int N, int K) {
  __shared__ unsigned short As[64][36];
  __shared__ unsigned short Bs[128][36];
  int tid = threadIdx.x, lane = tid & 31, wid = tid >> 5;
  int wr = (wid >> 1) * 32, wc = (wid & 1) * 64;
  int m0 = blockIdx.y * 64, n0 = blockIdx.x * 128;

  v8f acc[2][4];
  #pragma unroll
  for (int i = 0; i < 2; ++i)
    #pragma unroll
    for (int j = 0; j < 4; ++j) acc[i][j] = zero8();

  int arow = tid >> 3;          // 0..15
  int acol = (tid & 7) * 4;     // 0,4,..28

  float4 av[4], bv[8];
  auto load_stage = [&](int kk) {
    #pragma unroll
    for (int rr = 0; rr < 4; ++rr) {
      int gm = m0 + rr * 16 + arow;
      av[rr] = make_float4(0.f, 0.f, 0.f, 0.f);
      if (gm < M) av[rr] = *(const float4*)(A + (size_t)gm * K + kk + acol);
    }
    #pragma unroll
    for (int pp = 0; pp < 8; ++pp) {
      int gn = n0 + pp * 16 + arow;
      bv[pp] = make_float4(0.f, 0.f, 0.f, 0.f);
      if (gn < N) bv[pp] = *(const float4*)(W + (size_t)gn * K + kk + acol);
    }
  };
  auto store_stage = [&]() {
    #pragma unroll
    for (int rr = 0; rr < 4; ++rr)
      *(unsigned long long*)&As[rr * 16 + arow][acol] = f4_to_bf4(av[rr]);
    #pragma unroll
    for (int pp = 0; pp < 8; ++pp)
      *(unsigned long long*)&Bs[pp * 16 + arow][acol] = f4_to_bf4(bv[pp]);
  };

  load_stage(0);
  for (int kk = 0; kk < K; kk += 32) {
    store_stage();
    __syncthreads();
    if (kk + 32 < K) load_stage(kk + 32);   // overlap with WMMAs below

    v16bf afr[2], bfr[4];
    afr[0] = load_a_frag(As, wr, lane);
    afr[1] = load_a_frag(As, wr + 16, lane);
    #pragma unroll
    for (int ns = 0; ns < 4; ++ns) bfr[ns] = load_b_frag(Bs, wc + ns * 16, lane);
    #pragma unroll
    for (int ms = 0; ms < 2; ++ms)
      #pragma unroll
      for (int ns = 0; ns < 4; ++ns)
        acc[ms][ns] = __builtin_amdgcn_wmma_f32_16x16x32_bf16(
            false, afr[ms], false, bfr[ns], (short)0, acc[ms][ns], false, false);
    __syncthreads();
  }

  int hl = lane >> 4, nn = lane & 15;
  #pragma unroll
  for (int ms = 0; ms < 2; ++ms)
    #pragma unroll
    for (int ns = 0; ns < 4; ++ns)
      #pragma unroll
      for (int r = 0; r < 8; ++r) {
        int m = m0 + wr + ms * 16 + hl * 8 + r;
        int n = n0 + wc + ns * 16 + nn;
        if (m < M && n < N) {
          float v = acc[ms][ns][r];
          if (R) v += R[(size_t)m * N + n];
          C[(size_t)m * N + n] = v;
        }
      }
}

// ---------------- RoPE apply on q,k parts of qkv ----------------
__global__ void rope_apply_kernel(float* __restrict__ qkv,
                                  const float* __restrict__ cosT,
                                  const float* __restrict__ sinT) {
  int i = blockIdx.x * blockDim.x + threadIdx.x;
  if (i >= TOK * 512) return;       // rows * 2 parts * 16 heads * 16 freqs
  int f = i & 15;
  int head = (i >> 4) & 15;
  int part = (i >> 8) & 1;
  int row = i >> 9;
  int pos = row % SEQ;
  float c = cosT[pos * 16 + f], s = sinT[pos * 16 + f];
  size_t base = (size_t)row * QKVN + (size_t)part * DIMC + head * HD;
  float v1 = qkv[base + f], v2 = qkv[base + 16 + f];
  qkv[base + f]      = v1 * c - v2 * s;
  qkv[base + 16 + f] = v1 * s + v2 * c;
}

// ---------------- causal attention: one wave per (b, head, q) ----------------
__global__ __launch_bounds__(128) void attn_kernel(const float* __restrict__ qkv,
                                                   float* __restrict__ o) {
  __shared__ float sc[4][SEQ];
  int wid = threadIdx.x >> 5, lane = threadIdx.x & 31;
  int item = blockIdx.x * 4 + wid;          // 16416 items = 4104 blocks * 4 waves exactly
  int q = item % SEQ;
  int bh = item / SEQ;
  int h = bh & 15, b = bh >> 4;

  const float* qr = qkv + (size_t)(b * SEQ + q) * QKVN + h * HD;
  float qreg[HD];
  #pragma unroll
  for (int d = 0; d < HD; ++d) qreg[d] = qr[d];

  float mx = -1e30f;
  for (int k = lane; k <= q; k += 32) {
    const float* kr = qkv + (size_t)(b * SEQ + k) * QKVN + DIMC + h * HD;
    float s = 0.f;
    #pragma unroll 8
    for (int d = 0; d < HD; ++d) s += qreg[d] * kr[d];
    s *= 0.125f;                   // 1/sqrt(64)
    sc[wid][k] = s;
    mx = fmaxf(mx, s);
  }
  #pragma unroll
  for (int off = 16; off > 0; off >>= 1) mx = fmaxf(mx, __shfl_xor(mx, off, 32));
  __syncthreads();

  float sum = 0.f;
  for (int k = lane; k <= q; k += 32) {
    float p = expf(sc[wid][k] - mx);
    sc[wid][k] = p;
    sum += p;
  }
  #pragma unroll
  for (int off = 16; off > 0; off >>= 1) sum += __shfl_xor(sum, off, 32);
  __syncthreads();

  float o0 = 0.f, o1 = 0.f;
  for (int k = 0; k <= q; ++k) {
    float p = sc[wid][k];
    const float* vr = qkv + (size_t)(b * SEQ + k) * QKVN + 2 * DIMC + h * HD;
    o0 += p * vr[lane];
    o1 += p * vr[lane + 32];
  }
  float inv = 1.0f / sum;
  float* orow = o + (size_t)(b * SEQ + q) * DIMC + h * HD;
  orow[lane]      = o0 * inv;
  orow[lane + 32] = o1 * inv;
}

// ---------------- SwiGLU: ff = a * silu(g) ----------------
__global__ void glu_kernel(const float* __restrict__ up, float* __restrict__ ff) {
  int i = blockIdx.x * blockDim.x + threadIdx.x;
  if (i >= TOK * HID) return;
  int j = i % HID, t = i / HID;
  float a = up[(size_t)t * UPN + j];
  float g = up[(size_t)t * UPN + HID + j];
  ff[i] = a * g / (1.f + expf(-g));
}

// ---------------- router: softmax over 8 experts, top-2, compaction ----------------
__global__ void zero_cnt_kernel(int* __restrict__ cnt) {
  if (threadIdx.x < NEXP) cnt[threadIdx.x] = 0;
}

__global__ __launch_bounds__(32) void router_kernel(const float* __restrict__ xn,
                                                    const float* __restrict__ rw,
                                                    float* __restrict__ wk,
                                                    int* __restrict__ cnt,
                                                    int* __restrict__ lists) {
  int t = blockIdx.x, lane = threadIdx.x;
  float acc[NEXP];
  #pragma unroll
  for (int e = 0; e < NEXP; ++e) acc[e] = 0.f;
  const float* xr = xn + (size_t)t * DIMC;
  for (int k = lane; k < DIMC; k += 32) {
    float xv = xr[k];
    #pragma unroll
    for (int e = 0; e < NEXP; ++e) acc[e] += xv * rw[e * DIMC + k];
  }
  #pragma unroll
  for (int e = 0; e < NEXP; ++e)
    #pragma unroll
    for (int off = 16; off > 0; off >>= 1) acc[e] += __shfl_xor(acc[e], off, 32);

  if (lane == 0) {
    float mx = acc[0];
    #pragma unroll
    for (int e = 1; e < NEXP; ++e) mx = fmaxf(mx, acc[e]);
    float ex[NEXP], sum = 0.f;
    #pragma unroll
    for (int e = 0; e < NEXP; ++e) { ex[e] = expf(acc[e] - mx); sum += ex[e]; }
    int i0 = 0;
    #pragma unroll
    for (int e = 1; e < NEXP; ++e) if (ex[e] > ex[i0]) i0 = e;
    int i1 = (i0 == 0) ? 1 : 0;
    #pragma unroll
    for (int e = 0; e < NEXP; ++e) if (e != i0 && ex[e] > ex[i1]) i1 = e;
    float w0 = ex[i0] / sum, w1 = ex[i1] / sum;
    float ws = w0 + w1;
    wk[t * 2 + 0] = w0 / ws;
    wk[t * 2 + 1] = w1 / ws;
    int p0 = atomicAdd(&cnt[i0], 1);
    lists[i0 * MAXR + p0] = t * 2 + 0;
    int p1 = atomicAdd(&cnt[i1], 1);
    lists[i1 * MAXR + p1] = t * 2 + 1;
  }
}

// ---------------- expert GEMM (gathered rows, scattered stores) ----------------
__global__ __launch_bounds__(128) void expert_gemm_kernel(const float* __restrict__ xn,
                                                          const float* __restrict__ ew,
                                                          const int* __restrict__ cnt,
                                                          const int* __restrict__ lists,
                                                          float* __restrict__ out0,
                                                          float* __restrict__ out1) {
  int e = blockIdx.z;
  int Me = cnt[e];
  int m0 = blockIdx.y * 64;
  if (m0 >= Me) return;                 // uniform per-block

  __shared__ int rowmap[64];
  __shared__ unsigned short As[64][36];
  __shared__ unsigned short Bs[128][36];
  int tid = threadIdx.x, lane = tid & 31, wid = tid >> 5;
  int wr = (wid >> 1) * 32, wc = (wid & 1) * 64;
  int n0 = blockIdx.x * 128;

  if (tid < 64) {
    int gm = m0 + tid;
    rowmap[tid] = (gm < Me) ? lists[e * MAXR + gm] : -1;
  }
  __syncthreads();

  const float* W = ew + (size_t)e * VOC * DIMC;

  v8f acc[2][4];
  #pragma unroll
  for (int i = 0; i < 2; ++i)
    #pragma unroll
    for (int j = 0; j < 4; ++j) acc[i][j] = zero8();

  int arow = tid >> 3;
  int acol = (tid & 7) * 4;

  float4 av[4], bv[8];
  auto load_stage = [&](int kk) {
    #pragma unroll
    for (int rr = 0; rr < 4; ++rr) {
      int rm = rowmap[rr * 16 + arow];
      av[rr] = make_float4(0.f, 0.f, 0.f, 0.f);
      if (rm >= 0) av[rr] = *(const float4*)(xn + (size_t)(rm >> 1) * DIMC + kk + acol);
    }
    #pragma unroll
    for (int pp = 0; pp < 8; ++pp) {
      int gn = n0 + pp * 16 + arow;   // VOC multiple of 128 -> always in range
      bv[pp] = *(const float4*)(W + (size_t)gn * DIMC + kk + acol);
    }
  };
  auto store_stage = [&]() {
    #pragma unroll
    for (int rr = 0; rr < 4; ++rr)
      *(unsigned long long*)&As[rr * 16 + arow][acol] = f4_to_bf4(av[rr]);
    #pragma unroll
    for (int pp = 0; pp < 8; ++pp)
      *(unsigned long long*)&Bs[pp * 16 + arow][acol] = f4_to_bf4(bv[pp]);
  };

  load_stage(0);
  for (int kk = 0; kk < DIMC; kk += 32) {
    store_stage();
    __syncthreads();
    if (kk + 32 < DIMC) load_stage(kk + 32);

    v16bf afr[2], bfr[4];
    afr[0] = load_a_frag(As, wr, lane);
    afr[1] = load_a_frag(As, wr + 16, lane);
    #pragma unroll
    for (int ns = 0; ns < 4; ++ns) bfr[ns] = load_b_frag(Bs, wc + ns * 16, lane);
    #pragma unroll
    for (int ms = 0; ms < 2; ++ms)
      #pragma unroll
      for (int ns = 0; ns < 4; ++ns)
        acc[ms][ns] = __builtin_amdgcn_wmma_f32_16x16x32_bf16(
            false, afr[ms], false, bfr[ns], (short)0, acc[ms][ns], false, false);
    __syncthreads();
  }

  int hl = lane >> 4, nn = lane & 15;
  #pragma unroll
  for (int ms = 0; ms < 2; ++ms)
    #pragma unroll
    for (int ns = 0; ns < 4; ++ns)
      #pragma unroll
      for (int r = 0; r < 8; ++r) {
        int mi = wr + ms * 16 + hl * 8 + r;       // tile-local row
        int t2 = rowmap[mi];
        if (t2 >= 0) {
          int t = t2 >> 1, s = t2 & 1;
          float* base = s ? out1 : out0;
          base[(size_t)t * VOC + n0 + wc + ns * 16 + nn] = acc[ms][ns][r];
        }
      }
}

// ---------------- final: per-slot log_softmax + weighted logsumexp ----------------
__global__ __launch_bounds__(256) void combine_kernel(float* __restrict__ out0,
                                                      const float* __restrict__ in1,
                                                      const float* __restrict__ wkbuf) {
  __shared__ float red[8];
  int t = blockIdx.x, tid = threadIdx.x;
  float* L0 = out0 + (size_t)t * VOC;
  const float* L1 = in1 + (size_t)t * VOC;
  float w0 = wkbuf[t * 2 + 0], w1 = wkbuf[t * 2 + 1];

  float m0 = -3.0e38f, m1 = -3.0e38f;
  for (int v = tid; v < VOC; v += 256) {
    m0 = fmaxf(m0, L0[v]);
    m1 = fmaxf(m1, L1[v]);
  }
  m0 = blk_reduce_max(m0, red);
  m1 = blk_reduce_max(m1, red);

  float s0 = 0.f, s1 = 0.f;
  for (int v = tid; v < VOC; v += 256) {
    s0 += expf(L0[v] - m0);
    s1 += expf(L1[v] - m1);
  }
  s0 = blk_reduce_sum(s0, red);
  s1 = blk_reduce_sum(s1, red);
  float lse0 = m0 + logf(s0), lse1 = m1 + logf(s1);

  for (int v = tid; v < VOC; v += 256) {
    float a = w0 * expf(L0[v] - lse0) + w1 * expf(L1[v] - lse1);
    L0[v] = logf(a);
  }
}

// ---------------- host orchestration ----------------
extern "C" void kernel_launch(void* const* d_in, const int* in_sizes, int n_in,
                              void* d_out, int out_size, void* d_ws, size_t ws_size,
                              hipStream_t stream) {
  (void)in_sizes; (void)n_in; (void)out_size; (void)ws_size;

  const int* x = (const int*)d_in[0];
  const int* y = (const int*)d_in[1];
  const float* cls = (const float*)d_in[2];
  const float *attn_norm[6], *qkv_w[6], *attn_out_w[6], *ff_norm[6], *up_w[6], *down_w[6];
  for (int i = 0; i < 6; ++i) {
    int b = 3 + 6 * i;
    attn_norm[i]  = (const float*)d_in[b + 0];
    qkv_w[i]      = (const float*)d_in[b + 1];
    attn_out_w[i] = (const float*)d_in[b + 2];
    ff_norm[i]    = (const float*)d_in[b + 3];
    up_w[i]       = (const float*)d_in[b + 4];
    down_w[i]     = (const float*)d_in[b + 5];
  }
  const float* out_norm = (const float*)d_in[39];
  const float* router_w = (const float*)d_in[40];
  const float* expert_w = (const float*)d_in[41];
  float* out = (float*)d_out;

  char* wsb = (char*)d_ws;
  size_t off = 0;
  auto take = [&](size_t bytes) -> void* {
    void* p = wsb + off;
    off = (off + bytes + 255) & ~(size_t)255;
    return p;
  };
  float* cosT  = (float*)take((size_t)SEQ * 16 * 4);
  float* sinT  = (float*)take((size_t)SEQ * 16 * 4);
  float* h     = (float*)take((size_t)TOK * DIMC * 4);
  float* xn    = (float*)take((size_t)TOK * DIMC * 4);
  float* qkvb  = (float*)take((size_t)TOK * QKVN * 4);
  float* ao    = (float*)take((size_t)TOK * DIMC * 4);
  float* upb   = (float*)take((size_t)TOK * UPN * 4);
  float* ffb   = (float*)take((size_t)TOK * HID * 4);
  float* wk    = (float*)take((size_t)MAXR * 4);
  int*   cnt   = (int*)take((size_t)NEXP * 4);
  int*   lists = (int*)take((size_t)NEXP * MAXR * 4);
  float* slot1 = (float*)take((size_t)TOK * VOC * 4);

  rope_init_kernel<<<(SEQ * 16 + 255) / 256, 256, 0, stream>>>(cosT, sinT);
  embed_kernel<<<(TOK * DIMC + 255) / 256, 256, 0, stream>>>(x, y, cls, expert_w, h);

  for (int L = 0; L < 6; ++L) {
    rmsnorm_kernel<<<TOK, 256, 0, stream>>>(h, attn_norm[L], xn);
    gemm_kernel<<<dim3(QKVN / 128, 17), 128, 0, stream>>>(xn, qkv_w[L], nullptr, qkvb,
                                                          TOK, QKVN, DIMC);
    rope_apply_kernel<<<(TOK * 512 + 255) / 256, 256, 0, stream>>>(qkvb, cosT, sinT);
    attn_kernel<<<(2 * NH * SEQ) / 4, 128, 0, stream>>>(qkvb, ao);
    gemm_kernel<<<dim3(DIMC / 128, 17), 128, 0, stream>>>(ao, attn_out_w[L], h, h,
                                                          TOK, DIMC, DIMC);
    rmsnorm_kernel<<<TOK, 256, 0, stream>>>(h, ff_norm[L], xn);
    gemm_kernel<<<dim3(UPN / 128, 17), 128, 0, stream>>>(xn, up_w[L], nullptr, upb,
                                                         TOK, UPN, DIMC);
    glu_kernel<<<(TOK * HID + 255) / 256, 256, 0, stream>>>(upb, ffb);
    gemm_kernel<<<dim3(DIMC / 128, 17), 128, 0, stream>>>(ffb, down_w[L], h, h,
                                                          TOK, DIMC, HID);
  }

  rmsnorm_kernel<<<TOK, 256, 0, stream>>>(h, out_norm, xn);
  zero_cnt_kernel<<<1, 32, 0, stream>>>(cnt);
  router_kernel<<<TOK, 32, 0, stream>>>(xn, router_w, wk, cnt, lists);
  expert_gemm_kernel<<<dim3(VOC / 128, (MAXR + 63) / 64, NEXP), 128, 0, stream>>>(
      xn, expert_w, cnt, lists, out, slot1);
  combine_kernel<<<TOK, 256, 0, stream>>>(out, slot1, wk);
}
